// CompatiblePositionNet_69166153335442
// MI455X (gfx1250) — matrix operand order repaired and checked by reading, not compiled, not run on hardware
//
#include <hip/hip_runtime.h>
#include <math.h>

// ---------------------------------------------------------------------------
// Problem constants (from the reference)
// ---------------------------------------------------------------------------
#define BB     16
#define NN     512
#define EE     4096
#define IN_DIM 768
#define OUT_D  768
#define RELD   256
#define HEADS  8
#define HD     96          // OUT_D / HEADS
#define POSD   64
#define MTOK   (BB * NN)   // 8192

typedef __bf16 bf;
typedef __attribute__((ext_vector_type(8)))  bf    v8bf;
typedef __attribute__((ext_vector_type(16))) bf    v16bf;
typedef __attribute__((ext_vector_type(8)))  float v8f;
typedef unsigned int u32x4 __attribute__((ext_vector_type(4)));
typedef int          i32x4 __attribute__((ext_vector_type(4)));
typedef int          i32x8 __attribute__((ext_vector_type(8)));

#define DEV __device__ __forceinline__

// ---------------------------------------------------------------------------
// WMMA fragment helpers (CDNA5 16x16x32 bf16 layouts, CDNA5 ISA)
//   A (16x32, MxK): lane L holds row M=L%16, K = (L/16)*8 + [0..7] in v0..3
//                   and K = (L/16)*8 + 16 + [0..7] in v4..7
//   B (32x16, KxN): lane L holds col N=L%16, K same as A (table order)
//   C/D (16x16)   : vgpr i, lane L -> M = i + (L/16)*8, N = L%16
// ---------------------------------------------------------------------------
DEV v16bf fragA(const bf* tile, int ldk, int lane) {
  const bf* p = tile + (size_t)(lane & 15) * ldk + ((lane >> 4) << 3);
  v8bf lo = *(const v8bf*)p;
  v8bf hi = *(const v8bf*)(p + 16);
  return __builtin_shufflevector(lo, hi, 0,1,2,3,4,5,6,7,8,9,10,11,12,13,14,15);
}

DEV v16bf fragB(const bf* tile, int ldk, int lane) {
  const bf* p = tile + (size_t)(lane & 15) * ldk + ((lane >> 4) << 3);
  v8bf lo = *(const v8bf*)p;
  v8bf hi = *(const v8bf*)(p + 16);
  return __builtin_shufflevector(lo, hi, 0,1,2,3,4,5,6,7,8,9,10,11,12,13,14,15);
}

DEV v8f wmma_bf16(v16bf a, v16bf b, v8f c) {
  return __builtin_amdgcn_wmma_f32_16x16x32_bf16(false, a, false, b,
                                                 (short)0, c, false, false);
}

DEV float silu(float v) { return v / (1.f + __expf(-v)); }

// ---------------------------------------------------------------------------
// Tensor Data Mover: 2D tile (tile_k elems wide x tile_rows) of a bf16
// matrix with row stride `stride_elems`, DMA'd global -> LDS at lds_off.
// D# layout per CDNA5 ISA (data_size=1 -> 2 bytes).
// ---------------------------------------------------------------------------
DEV void tdm_load_2d(unsigned lds_off, const void* gsrc, unsigned tile_k,
                     unsigned tile_rows, unsigned stride_elems) {
  unsigned long long ga = (unsigned long long)(uintptr_t)gsrc;
  unsigned td0 = stride_elems;      // tensor_dim0 (>= tile_k, no clipping)
  unsigned td1 = 0x00100000u;       // tensor_dim1: huge, no clipping
  u32x4 g0;
  g0[0] = 1u;                                   // count=1, user descriptor
  g0[1] = lds_off;                              // lds_addr (bytes)
  g0[2] = (unsigned)ga;                         // global_addr[31:0]
  g0[3] = (unsigned)(ga >> 32) | (2u << 30);    // global_addr[56:32] | type=2
  i32x8 g1;
  g1[0] = (int)(1u << 16);                      // data_size=1 (2 bytes)
  g1[1] = (int)((td0 & 0xFFFFu) << 16);         // tensor_dim0[15:0]
  g1[2] = (int)((td0 >> 16) | ((td1 & 0xFFFFu) << 16)); // td0[31:16],td1[15:0]
  g1[3] = (int)((td1 >> 16) | (tile_k << 16));  // td1[31:16], tile_dim0
  g1[4] = (int)(tile_rows & 0xFFFFu);           // tile_dim1, tile_dim2=0
  g1[5] = (int)stride_elems;                    // tensor_dim0_stride[31:0]
  g1[6] = 0;                                    // stride[47:32], d1_stride lo
  g1[7] = 0;
  i32x4 z4 = {0, 0, 0, 0};
#if __clang_major__ >= 23
  i32x8 z8 = {0, 0, 0, 0, 0, 0, 0, 0};
  __builtin_amdgcn_tensor_load_to_lds(g0, g1, z4, z4, z8, 0);
#else
  __builtin_amdgcn_tensor_load_to_lds(g0, g1, z4, z4, 0);
#endif
}

// ---------------------------------------------------------------------------
// Shared GEMM epilogue.
// MODE 0: SiLU -> bf16 linear          MODE 1: *mask[row] -> f32 linear
// MODE 2: -> bf16 [B,H,N,HD] (Q,K)     MODE 3: -> bf16 [B,H,HD,N] (V^T)
// MODE 4: += into f32 linear (resid)   MODE 6: (f32+=v)*mask[row] (FF resid)
// ---------------------------------------------------------------------------
template <int MODE>
DEV void gemm_store(v8f a00, v8f a01, v8f a10, v8f a11, int m0, int n0, int N,
                    int lane, const float* __restrict__ bias,
                    bf* __restrict__ outb, float* __restrict__ outf,
                    const float* __restrict__ mask) {
  v8f accs[4] = {a00, a01, a10, a11};
  #pragma unroll
  for (int tile = 0; tile < 4; ++tile) {
    int im = tile >> 1, in = tile & 1;
    int mbase = m0 + im * 16 + (lane >> 4) * 8;
    int nb = n0 + in * 16 + (lane & 15);
    float bv = bias[nb];
    #pragma unroll
    for (int i = 0; i < 8; ++i) {
      int m = mbase + i;
      float v = accs[tile][i] + bv;
      if (MODE == 0) {
        outb[(size_t)m * N + nb] = (bf)silu(v);
      } else if (MODE == 1) {
        outf[(size_t)m * N + nb] = v * mask[m];
      } else if (MODE == 2) {
        int b = m >> 9, t = m & 511, h = nb / HD, d = nb % HD;
        outb[(((size_t)(b * HEADS + h) * NN) + t) * HD + d] = (bf)v;
      } else if (MODE == 3) {
        int b = m >> 9, t = m & 511, h = nb / HD, d = nb % HD;
        outb[(((size_t)(b * HEADS + h) * HD) + d) * NN + t] = (bf)v;
      } else if (MODE == 4) {
        outf[(size_t)m * N + nb] += v;
      } else {  // MODE 6
        size_t idx = (size_t)m * N + nb;
        outf[idx] = (outf[idx] + v) * mask[m];
      }
    }
  }
}

// ---------------------------------------------------------------------------
// f32 -> bf16 conversion
// ---------------------------------------------------------------------------
__global__ void f32_to_bf16_k(const float* __restrict__ s, bf* __restrict__ d,
                              size_t n) {
  size_t i = (size_t)blockIdx.x * blockDim.x + threadIdx.x;
  if (i < n) d[i] = (bf)s[i];
}

// ---------------------------------------------------------------------------
// Fourier embed + concat -> X0 [MTOK, 832] bf16
// ---------------------------------------------------------------------------
__global__ void build_x0_k(const float* __restrict__ pos,
                           const float* __restrict__ boxes,
                           bf* __restrict__ x0) {
  int t = blockIdx.x;  // token
  for (int c = threadIdx.x; c < IN_DIM + POSD; c += blockDim.x) {
    float v;
    if (c < IN_DIM) {
      v = pos[(size_t)t * IN_DIM + c];
    } else {
      int j = c - IN_DIM;
      int f = j >> 3, s = (j >> 2) & 1, cc = j & 3;
      float freq = __powf(100.f, (float)f * 0.125f);
      float xv = boxes[(size_t)t * 4 + cc] * freq;
      v = s ? __cosf(xv) : __sinf(xv);
    }
    x0[(size_t)t * (IN_DIM + POSD) + c] = (bf)v;
  }
}

// ---------------------------------------------------------------------------
// Generic WMMA GEMM (B from global):  out = epi( X[M,K] @ W[N,K]^T + bias )
// block = 256 threads = 8 waves; block tile 128x64; wave tile 32x32.
// ---------------------------------------------------------------------------
template <int MODE>
__global__ __launch_bounds__(256)
void gemm_bf16_k(const bf* __restrict__ X, const bf* __restrict__ W,
                 const float* __restrict__ bias, int M, int N, int K,
                 bf* __restrict__ outb, float* __restrict__ outf,
                 const float* __restrict__ mask) {
  int nbn = N >> 6;
  int bm = blockIdx.x / nbn, bn = blockIdx.x % nbn;
  int w = threadIdx.x >> 5, lane = threadIdx.x & 31;
  int m0 = bm * 128 + (w >> 1) * 32;
  int n0 = bn * 64 + (w & 1) * 32;

  v8f acc00 = {}, acc01 = {}, acc10 = {}, acc11 = {};
  for (int k0 = 0; k0 < K; k0 += 32) {
    if (k0 + 32 < K) {  // global_prefetch_b8 hint for next K-slab
      __builtin_prefetch(X + (size_t)m0 * K + k0 + 32, 0, 1);
      __builtin_prefetch(W + (size_t)n0 * K + k0 + 32, 0, 1);
    }
    v16bf a0 = fragA(X + (size_t)m0 * K + k0, K, lane);
    v16bf a1 = fragA(X + (size_t)(m0 + 16) * K + k0, K, lane);
    v16bf b0 = fragB(W + (size_t)n0 * K + k0, K, lane);
    v16bf b1 = fragB(W + (size_t)(n0 + 16) * K + k0, K, lane);
    acc00 = wmma_bf16(a0, b0, acc00);
    acc01 = wmma_bf16(a0, b1, acc01);
    acc10 = wmma_bf16(a1, b0, acc10);
    acc11 = wmma_bf16(a1, b1, acc11);
  }
  gemm_store<MODE>(acc00, acc01, acc10, acc11, m0, n0, N, lane, bias, outb,
                   outf, mask);
}

// ---------------------------------------------------------------------------
// TDM-staged WMMA GEMM (K % 128 == 0): wave 0 DMA's the 64x128 weight slab
// into LDS via tensor_load_to_lds (TENSORcnt), all 8 waves consume B
// fragments from LDS; A streams from global.
// ---------------------------------------------------------------------------
template <int MODE>
__global__ __launch_bounds__(256)
void gemm_bf16_tdm_k(const bf* __restrict__ X, const bf* __restrict__ W,
                     const float* __restrict__ bias, int M, int N, int K,
                     bf* __restrict__ outb, float* __restrict__ outf,
                     const float* __restrict__ mask) {
  __shared__ bf wtile[64][128];  // 16 KB weight slab
  int nbn = N >> 6;
  int bm = blockIdx.x / nbn, bn = blockIdx.x % nbn;
  int w = threadIdx.x >> 5, lane = threadIdx.x & 31;
  int m0 = bm * 128 + (w >> 1) * 32;
  int nw = (w & 1) * 32;         // n offset inside the 64-wide block tile
  int n0 = bn * 64 + nw;
  unsigned lds_off = (unsigned)(uintptr_t)(void*)&wtile[0][0];

  v8f acc00 = {}, acc01 = {}, acc10 = {}, acc11 = {};
  for (int ks = 0; ks < K; ks += 128) {
    __syncthreads();  // previous slab fully consumed
    if (w == 0) {
      tdm_load_2d(lds_off, W + (size_t)(bn * 64) * K + ks, 128u, 64u,
                  (unsigned)K);
      __builtin_amdgcn_s_wait_tensorcnt(0);
    }
    __syncthreads();  // slab visible to all waves
    #pragma unroll
    for (int kk = 0; kk < 128; kk += 32) {
      v16bf a0 = fragA(X + (size_t)m0 * K + ks + kk, K, lane);
      v16bf a1 = fragA(X + (size_t)(m0 + 16) * K + ks + kk, K, lane);
      v16bf b0 = fragB(&wtile[nw][0] + kk, 128, lane);
      v16bf b1 = fragB(&wtile[nw + 16][0] + kk, 128, lane);
      acc00 = wmma_bf16(a0, b0, acc00);
      acc01 = wmma_bf16(a0, b1, acc01);
      acc10 = wmma_bf16(a1, b0, acc10);
      acc11 = wmma_bf16(a1, b1, acc11);
    }
  }
  gemm_store<MODE>(acc00, acc01, acc10, acc11, m0, n0, N, lane, bias, outb,
                   outf, mask);
}

// ---------------------------------------------------------------------------
// LayerNorm over last dim (768), f32 in -> bf16 out. 1 wave per row.
// ---------------------------------------------------------------------------
__global__ __launch_bounds__(256)
void layernorm_k(const float* __restrict__ x, const float* __restrict__ g,
                 const float* __restrict__ bta, bf* __restrict__ out) {
  int w = threadIdx.x >> 5, lane = threadIdx.x & 31;
  int row = blockIdx.x * 8 + w;
  const float* xr = x + (size_t)row * OUT_D;
  float vals[OUT_D / 32];
  float s = 0.f;
  #pragma unroll
  for (int i = 0; i < OUT_D / 32; ++i) { vals[i] = xr[lane + i * 32]; s += vals[i]; }
  #pragma unroll
  for (int o = 16; o >= 1; o >>= 1) s += __shfl_xor(s, o, 32);
  float mean = s * (1.f / OUT_D);
  float vs = 0.f;
  #pragma unroll
  for (int i = 0; i < OUT_D / 32; ++i) { float d = vals[i] - mean; vs += d * d; }
  #pragma unroll
  for (int o = 16; o >= 1; o >>= 1) vs += __shfl_xor(vs, o, 32);
  float inv = rsqrtf(vs * (1.f / OUT_D) + 1e-5f);
  #pragma unroll
  for (int i = 0; i < OUT_D / 32; ++i) {
    int c = lane + i * 32;
    out[(size_t)row * OUT_D + c] = (bf)((vals[i] - mean) * inv * g[c] + bta[c]);
  }
}

// ---------------------------------------------------------------------------
// Per-edge per-head bias: eb[b,e,h] = valid ? rel[b,e,:]·We[h,:] + be[h] : 0
// ---------------------------------------------------------------------------
__global__ __launch_bounds__(256)
void edge_bias_k(const float* __restrict__ rel, const float* __restrict__ rmask,
                 const float* __restrict__ We, const float* __restrict__ be,
                 float* __restrict__ eb) {
  int e = blockIdx.x * 32 + (threadIdx.x >> 3);
  int h = threadIdx.x & 7;
  const float* r = rel + (size_t)e * RELD;
  const float* wv = We + (size_t)h * RELD;
  float s = be[h];
  for (int k = 0; k < RELD; ++k) s += r[k] * wv[k];
  eb[(size_t)e * HEADS + h] = (rmask[e] > 0.f) ? s : 0.f;
}

// ---------------------------------------------------------------------------
// Dense per-batch edge table: table[b,q,k] = edge id (atomicMax = det. winner)
// ---------------------------------------------------------------------------
__global__ void scatter_edges_k(const int* __restrict__ edges,
                                const float* __restrict__ rmask,
                                int* __restrict__ table) {
  int e = blockIdx.x * 256 + threadIdx.x;
  if (e >= BB * EE) return;
  if (rmask[e] <= 0.f) return;
  int b = e >> 12, el = e & (EE - 1);
  int s = edges[(size_t)e * 2];     s = s < 0 ? 0 : (s > NN - 1 ? NN - 1 : s);
  int d = edges[(size_t)e * 2 + 1]; d = d < 0 ? 0 : (d > NN - 1 ? NN - 1 : d);
  atomicMax(&table[((size_t)b * NN + s) * NN + d], el);
  atomicMax(&table[((size_t)b * NN + d) * NN + s], el);
}

// ---------------------------------------------------------------------------
// Attention: one block per (b, h, 16-row q-block). 256 thr = 8 waves.
//  Phase 1: S = Q K^T * scale + bias, masked -> LDS (WMMA, hd in 3x32 chunks)
//  Phase 2: exact softmax in LDS (nan_to_num guard) -> bf16 P in LDS
//  Phase 3: O = P @ V (K-dim 512 in 16x32 chunks) -> Xout[b,t,h*96+d] bf16
// ---------------------------------------------------------------------------
__global__ __launch_bounds__(256)
void attention_k(const bf* __restrict__ Q, const bf* __restrict__ Km,
                 const bf* __restrict__ Vt, const float* __restrict__ eb,
                 const int* __restrict__ table, const float* __restrict__ masks,
                 bf* __restrict__ Xout) {
  __shared__ float S[16][NN];
  __shared__ bf    P[16][NN];
  __shared__ float red[16][16];
  __shared__ float rstat[16];

  int bh = blockIdx.x >> 5;      // b*8 + h
  int qb = blockIdx.x & 31;
  int b = bh >> 3, h = bh & 7;
  int w = threadIdx.x >> 5, lane = threadIdx.x & 31;
  const bf* Qs = Q + (size_t)bh * NN * HD;
  const bf* Ks = Km + (size_t)bh * NN * HD;
  const bf* Vs = Vt + (size_t)bh * HD * NN;
  int q0 = qb * 16;
  const float scale = 0.1020620726159658f;  // 1/sqrt(96)

  // ---- phase 1: scores
  for (int kt = w; kt < 32; kt += 8) {
    v8f acc = {};
    #pragma unroll
    for (int c = 0; c < HD; c += 32) {
      v16bf a  = fragA(Qs + (size_t)q0 * HD + c, HD, lane);
      v16bf bb = fragB(Ks + (size_t)(kt * 16) * HD + c, HD, lane);
      acc = wmma_bf16(a, bb, acc);
    }
    int kk = kt * 16 + (lane & 15);
    bool nmk = masks[b * NN + kk] > 0.f;
    #pragma unroll
    for (int i = 0; i < 8; ++i) {
      int qloc = (lane >> 4) * 8 + i;
      int qq = q0 + qloc;
      int e = table[((size_t)b * NN + qq) * NN + kk];
      bool nmq = masks[b * NN + qq] > 0.f;
      bool ok = nmq && nmk && ((qq == kk) || (e >= 0));
      float v = acc[i] * scale +
                ((e >= 0) ? eb[((size_t)b * EE + e) * HEADS + h] : 0.f);
      S[qloc][kk] = ok ? v : -3.4e38f;
    }
  }
  __syncthreads();

  // ---- phase 2: softmax (16 threads per row)
  {
    int row = threadIdx.x >> 4, sub = threadIdx.x & 15;
    float mx = -3.4e38f;
    for (int k = sub; k < NN; k += 16) mx = fmaxf(mx, S[row][k]);
    red[row][sub] = mx;
    __syncthreads();
    if (sub == 0) {
      float m2 = red[row][0];
      #pragma unroll
      for (int i = 1; i < 16; ++i) m2 = fmaxf(m2, red[row][i]);
      rstat[row] = m2;
    }
    __syncthreads();
    float rmax = rstat[row];
    float sum = 0.f;
    for (int k = sub; k < NN; k += 16) {
      float p = (rmax > -1e37f) ? __expf(S[row][k] - rmax) : 0.f;
      S[row][k] = p;
      sum += p;
    }
    red[row][sub] = sum;
    __syncthreads();
    if (sub == 0) {
      float s2 = 0.f;
      #pragma unroll
      for (int i = 0; i < 16; ++i) s2 += red[row][i];
      rstat[row] = (s2 > 0.f) ? 1.f / s2 : 0.f;  // nan_to_num
    }
    __syncthreads();
    float inv = rstat[row];
    for (int k = sub; k < NN; k += 16) P[row][k] = (bf)(S[row][k] * inv);
  }
  __syncthreads();

  // ---- phase 3: O = P @ V  (6 of 8 waves, one 16x16 hd-tile each)
  if (w < 6) {
    int n0 = w * 16;
    v8f acc = {};
    for (int kc = 0; kc < NN; kc += 32) {
      v16bf a  = fragA((const bf*)&P[0][0] + kc, NN, lane);
      v16bf bb = fragB(Vs + (size_t)n0 * NN + kc, NN, lane);
      acc = wmma_bf16(a, bb, acc);
    }
    int n = n0 + (lane & 15);
    #pragma unroll
    for (int i = 0; i < 8; ++i) {
      int t = q0 + (lane >> 4) * 8 + i;
      Xout[((size_t)(b * NN) + t) * OUT_D + h * HD + n] = (bf)acc[i];
    }
  }
}

// ---------------------------------------------------------------------------
// Final: out = objs + sigmoid(gate) * g
// ---------------------------------------------------------------------------
__global__ void final_k(const float* __restrict__ objs, const float* __restrict__ g,
                        const float* __restrict__ gate, float* __restrict__ out,
                        size_t n) {
  size_t i = (size_t)blockIdx.x * blockDim.x + threadIdx.x;
  if (i >= n) return;
  float sg = 1.f / (1.f + __expf(-gate[0]));
  out[i] = objs[i] + sg * g[i];
}

// ---------------------------------------------------------------------------
// Host side
// ---------------------------------------------------------------------------
template <int MODE>
static void launch_gemm(const bf* X, const bf* W, const float* bias, int M,
                        int N, int K, bf* ob, float* of, const float* mask,
                        hipStream_t s) {
  dim3 grid((unsigned)((M / 128) * (N / 64)));
  if ((K & 127) == 0)
    gemm_bf16_tdm_k<MODE><<<grid, 256, 0, s>>>(X, W, bias, M, N, K, ob, of, mask);
  else
    gemm_bf16_k<MODE><<<grid, 256, 0, s>>>(X, W, bias, M, N, K, ob, of, mask);
}

struct GatP {
  const float *W1, *W2, *We, *Wk, *Wo, *Wq, *Wv, *b1, *b2, *be, *bk, *bo, *bq,
      *bv, *ffln_b, *ffln_g, *ln_b, *ln_g;
};

extern "C" void kernel_launch(void* const* d_in, const int* in_sizes, int n_in,
                              void* d_out, int out_size, void* d_ws,
                              size_t ws_size, hipStream_t stream) {
  (void)in_sizes; (void)n_in; (void)ws_size;

  // Data inputs (setup_inputs() order)
  const float* boxes   = (const float*)d_in[0];
  const float* masks   = (const float*)d_in[1];
  const float* posemb  = (const float*)d_in[2];
  const int*   edges   = (const int*)d_in[3];
  const float* relmask = (const float*)d_in[4];
  const float* relemb  = (const float*)d_in[5];
  // Params: jax pytree (sorted-key) leaf order appended after the 6 arrays:
  //   gat0[18], gat1[18], gate, l0_W, l0_b, l1_W, l1_b, l2_W, l2_b
  //   gat leaves (sorted): W1,W2,We,Wk,Wo,Wq,Wv,b1,b2,be,bk,bo,bq,bv,
  //                        ffln_b,ffln_g,ln_b,ln_g
  auto getGat = [&](int base) {
    GatP g;
    g.W1 = (const float*)d_in[base + 0];  g.W2 = (const float*)d_in[base + 1];
    g.We = (const float*)d_in[base + 2];  g.Wk = (const float*)d_in[base + 3];
    g.Wo = (const float*)d_in[base + 4];  g.Wq = (const float*)d_in[base + 5];
    g.Wv = (const float*)d_in[base + 6];  g.b1 = (const float*)d_in[base + 7];
    g.b2 = (const float*)d_in[base + 8];  g.be = (const float*)d_in[base + 9];
    g.bk = (const float*)d_in[base + 10]; g.bo = (const float*)d_in[base + 11];
    g.bq = (const float*)d_in[base + 12]; g.bv = (const float*)d_in[base + 13];
    g.ffln_b = (const float*)d_in[base + 14];
    g.ffln_g = (const float*)d_in[base + 15];
    g.ln_b = (const float*)d_in[base + 16];
    g.ln_g = (const float*)d_in[base + 17];
    return g;
  };
  GatP gat[2] = {getGat(6), getGat(24)};
  const float* gate = (const float*)d_in[42];
  const float* l0W = (const float*)d_in[43]; const float* l0b = (const float*)d_in[44];
  const float* l1W = (const float*)d_in[45]; const float* l1b = (const float*)d_in[46];
  const float* l2W = (const float*)d_in[47]; const float* l2b = (const float*)d_in[48];

  // ---- workspace carve (256B aligned)
  size_t off = 0;
  auto carve = [&](size_t bytes) -> char* {
    off = (off + 255) & ~(size_t)255;
    char* p = (char*)d_ws + off;
    off += bytes;
    return p;
  };
  bf* wl0 = (bf*)carve((size_t)512 * 832 * 2);
  bf* wl1 = (bf*)carve((size_t)512 * 512 * 2);
  bf* wl2 = (bf*)carve((size_t)768 * 512 * 2);
  bf* gw[2][6];  // q,k,v,o,1,2
  for (int l = 0; l < 2; ++l) {
    for (int j = 0; j < 4; ++j) gw[l][j] = (bf*)carve((size_t)768 * 768 * 2);
    gw[l][4] = (bf*)carve((size_t)3072 * 768 * 2);
    gw[l][5] = (bf*)carve((size_t)768 * 3072 * 2);
  }
  bf*    x0    = (bf*)carve((size_t)MTOK * 832 * 2);
  bf*    h1    = (bf*)carve((size_t)MTOK * 512 * 2);
  bf*    h2    = (bf*)carve((size_t)MTOK * 512 * 2);
  float* objs  = (float*)carve((size_t)MTOK * OUT_D * 4);
  float* g     = (float*)carve((size_t)MTOK * OUT_D * 4);
  bf*    xn    = (bf*)carve((size_t)MTOK * OUT_D * 2);
  bf*    qm    = (bf*)carve((size_t)MTOK * OUT_D * 2);
  bf*    km    = (bf*)carve((size_t)MTOK * OUT_D * 2);
  bf*    vt    = (bf*)carve((size_t)MTOK * OUT_D * 2);
  float* ebuf  = (float*)carve((size_t)BB * EE * HEADS * 4);
  int*   table = (int*)carve((size_t)BB * NN * NN * 4);
  bf*    xat   = (bf*)carve((size_t)MTOK * OUT_D * 2);
  bf*    hff   = (bf*)carve((size_t)MTOK * 3072 * 2);

  auto conv = [&](const float* src, bf* dst, size_t n) {
    f32_to_bf16_k<<<(unsigned)((n + 255) / 256), 256, 0, stream>>>(src, dst, n);
  };

  // ---- weight conversion
  conv(l0W, wl0, (size_t)512 * 832);
  conv(l1W, wl1, (size_t)512 * 512);
  conv(l2W, wl2, (size_t)768 * 512);
  for (int l = 0; l < 2; ++l) {
    conv(gat[l].Wq, gw[l][0], (size_t)768 * 768);
    conv(gat[l].Wk, gw[l][1], (size_t)768 * 768);
    conv(gat[l].Wv, gw[l][2], (size_t)768 * 768);
    conv(gat[l].Wo, gw[l][3], (size_t)768 * 768);
    conv(gat[l].W1, gw[l][4], (size_t)3072 * 768);
    conv(gat[l].W2, gw[l][5], (size_t)768 * 3072);
  }

  // ---- trunk MLP
  build_x0_k<<<MTOK, 256, 0, stream>>>(posemb, boxes, x0);
  launch_gemm<0>(x0, wl0, l0b, MTOK, 512, 832, h1, nullptr, nullptr, stream);
  launch_gemm<0>(h1, wl1, l1b, MTOK, 512, 512, h2, nullptr, nullptr, stream);
  launch_gemm<1>(h2, wl2, l2b, MTOK, OUT_D, 512, nullptr, objs, masks, stream);
  hipMemcpyAsync(g, objs, (size_t)MTOK * OUT_D * 4, hipMemcpyDeviceToDevice,
                 stream);

  // ---- edge table (layer-independent)
  hipMemsetAsync(table, 0xFF, (size_t)BB * NN * NN * 4, stream);
  scatter_edges_k<<<(BB * EE) / 256, 256, 0, stream>>>(edges, relmask, table);

  // ---- GAT layers
  for (int l = 0; l < 2; ++l) {
    const GatP& p = gat[l];
    layernorm_k<<<MTOK / 8, 256, 0, stream>>>(g, p.ln_g, p.ln_b, xn);
    launch_gemm<2>(xn, gw[l][0], p.bq, MTOK, OUT_D, OUT_D, qm, nullptr, nullptr, stream);
    launch_gemm<2>(xn, gw[l][1], p.bk, MTOK, OUT_D, OUT_D, km, nullptr, nullptr, stream);
    launch_gemm<3>(xn, gw[l][2], p.bv, MTOK, OUT_D, OUT_D, vt, nullptr, nullptr, stream);
    edge_bias_k<<<(BB * EE) / 32, 256, 0, stream>>>(relemb, relmask, p.We, p.be, ebuf);
    attention_k<<<BB * HEADS * (NN / 16), 256, 0, stream>>>(qm, km, vt, ebuf,
                                                            table, masks, xat);
    launch_gemm<4>(xat, gw[l][3], p.bo, MTOK, OUT_D, OUT_D, nullptr, g, nullptr, stream);
    layernorm_k<<<MTOK / 8, 256, 0, stream>>>(g, p.ffln_g, p.ffln_b, xn);
    launch_gemm<0>(xn, gw[l][4], p.b1, MTOK, 3072, OUT_D, hff, nullptr, nullptr, stream);
    launch_gemm<6>(hff, gw[l][5], p.b2, MTOK, OUT_D, 3072, nullptr, g, masks, stream);
  }

  // ---- combine
  size_t n = (size_t)out_size;
  final_k<<<(unsigned)((n + 255) / 256), 256, 0, stream>>>(objs, g, gate,
                                                           (float*)d_out, n);
}
